// CausalTransformer_v2_21603685499547
// MI455X (gfx1250) — matrix operand order repaired and checked by reading, not compiled, not run on hardware
//
#include <hip/hip_runtime.h>
#include <math.h>
#include <stdint.h>

// ---------------------------------------------------------------------------
// Model constants (from reference): B=4, N=1024, M=512, D=512, P=256, H=8,
// DH=64, L=6, FI=2048, ROT_DIM=32, NUM_BUCKETS=32, MAX_DIST=128.
// ---------------------------------------------------------------------------
#define BB   4
#define NN   1024
#define MMCT 512     // context length M
#define DD   512
#define PPC  256     // context dim P
#define HHD  8       // heads
#define DHH  64      // head dim
#define IDM  512     // H*DH
#define LLY  6
#define FFI  2048

typedef __attribute__((ext_vector_type(16))) _Float16 v16h;
typedef __attribute__((ext_vector_type(8)))  _Float16 v8h;
typedef __attribute__((ext_vector_type(8)))  float    v8f;

static __device__ __forceinline__ v8f wmma_f16(v16h a, v16h b, v8f c) {
  // v_wmma_f32_16x16x32_f16  (wave32, 16x16 f32 accum = 8 VGPRs)
  return __builtin_amdgcn_wmma_f32_16x16x32_f16(
      /*neg_a=*/false, a, /*neg_b=*/false, b,
      /*c_mod=*/(short)0, c, /*reuse_a=*/false, /*reuse_b=*/false);
}
static __device__ __forceinline__ v8f zero8() {
  v8f z;
  #pragma unroll
  for (int e = 0; e < 8; ++e) z[e] = 0.0f;
  return z;
}

// Async DMA of one 16B chunk global->LDS (GLOBAL_LOAD_ASYNC_TO_LDS_B128,
// tracked on ASYNCcnt).  LDS byte address = low 32 bits of generic pointer
// (ISA aperture layout: the LDS aperture keeps the offset in addr[31:0]).
static __device__ __forceinline__ void async_copy16(const _Float16* lds_dst,
                                                    const _Float16* gsrc) {
  unsigned lds_off = (unsigned)(uintptr_t)lds_dst;
  unsigned long long ga = (unsigned long long)(uintptr_t)gsrc;
  asm volatile("global_load_async_to_lds_b128 %0, %1, off"
               :: "v"(lds_off), "v"(ga) : "memory");
}
static __device__ __forceinline__ void wait_async_le4() {
  asm volatile("s_wait_asynccnt 0x4" ::: "memory");
}
static __device__ __forceinline__ void wait_async0() {
  asm volatile("s_wait_asynccnt 0x0" ::: "memory");
}

// ---------------------------------------------------------------------------
// GEMM: C[Mr,NCOLS] = A[Mr,K] @ W[K,NCOLS]; A f16 row-major, W given
// TRANSPOSED as Wt[NCOLS][K] f16, C f32.  REQUIRES Mr % 128 == 0, K % 32 == 0.
// Block tile 128x128, 8 waves (4x2), wave tile 32x64 (2x4 16x16 accums).
// Both tiles staged by double-buffered async DMA (4 chunks/thread/tile):
// issue tile k+1, s_wait_asynccnt<=4 completes tile k (in-order), compute.
// ---------------------------------------------------------------------------
#define GBM 128
#define GBN 128
#define GBK 32
#define GLDA 40
#define GLDB 40

template <int NCOLS>
__global__ __launch_bounds__(256) void gemm_kernel(
    const _Float16* __restrict__ A, const _Float16* __restrict__ Wt,
    float* __restrict__ C, int Mr, int K)
{
  __shared__ __attribute__((aligned(16))) _Float16 As[2][GBM * GLDA];
  __shared__ __attribute__((aligned(16))) _Float16 Bs[2][GBN * GLDB];
  const int tid  = threadIdx.x;
  const int lane = tid & 31;
  const int wave = tid >> 5;
  const int wm = wave >> 1;      // 0..3  -> 32 rows each
  const int wn = wave & 1;       // 0..1  -> 64 cols each
  const int l16 = lane & 15;
  const int lhi = lane >> 4;
  const int row0 = blockIdx.y * GBM;
  const int col0 = blockIdx.x * GBN;

  v8f acc[2][4];
  #pragma unroll
  for (int i = 0; i < 2; ++i)
    #pragma unroll
    for (int j = 0; j < 4; ++j)
      acc[i][j] = zero8();

  // stage one 128x32 tile pair into buffer `buf` (4 async chunks / thread)
  auto issue = [&](int kt, int buf) {
    int k0 = kt * GBK;
    #pragma unroll
    for (int j = 0; j < 2; ++j) {
      int c = tid + 256 * j;          // 512 chunks of 8 halves
      int r = c >> 2, c8 = c & 3;     // 4 chunks per 32-half row
      async_copy16(&As[buf][r * GLDA + c8 * 8],
                   &A[(size_t)(row0 + r) * K + k0 + c8 * 8]);
      async_copy16(&Bs[buf][r * GLDB + c8 * 8],
                   &Wt[(size_t)(col0 + r) * K + k0 + c8 * 8]);
    }
  };

  const int KT = K / GBK;
  issue(0, 0);
  for (int kt = 0; kt < KT; ++kt) {
    const int cur = kt & 1;
    const bool more = (kt + 1 < KT);
    if (more) issue(kt + 1, cur ^ 1);   // prefetch next tile into other buffer
    if (more) wait_async_le4();          // tile kt's 4 chunks complete
    else      wait_async0();
    __syncthreads();                     // all waves' staging of `cur` visible

    v16h af[2], bf[4];
    #pragma unroll
    for (int i = 0; i < 2; ++i) {
      const _Float16* base = &As[cur][(wm * 32 + i * 16 + l16) * GLDA];
      #pragma unroll
      for (int e = 0; e < 16; ++e)
        af[i][e] = base[lhi * 16 + e];
    }
    #pragma unroll
    for (int j = 0; j < 4; ++j) {
      const _Float16* base = &Bs[cur][(wn * 64 + j * 16 + l16) * GLDB + lhi * 16];
      #pragma unroll
      for (int e = 0; e < 16; ++e) bf[j][e] = base[e];
    }
    #pragma unroll
    for (int i = 0; i < 2; ++i)
      #pragma unroll
      for (int j = 0; j < 4; ++j)
        acc[i][j] = wmma_f16(af[i], bf[j], acc[i][j]);
    __syncthreads();   // reads of `cur` done before it is re-targeted
  }

  #pragma unroll
  for (int i = 0; i < 2; ++i)
    #pragma unroll
    for (int j = 0; j < 4; ++j) {
      int col = col0 + wn * 64 + j * 16 + l16;
      #pragma unroll
      for (int v = 0; v < 8; ++v) {
        int row = row0 + wm * 32 + i * 16 + lhi * 8 + v;
        C[(size_t)row * NCOLS + col] = acc[i][j][v];
      }
    }
}

// ---------------------------------------------------------------------------
// Flash-style attention, all-f16 operands: per block 4 waves x 16 query rows,
// streaming over 32-key tiles. Qh:(B*Nq,512); Kc:(B,Lk,64); Vt:(B,64,Lkp)
// (V transposed, padded key stride).  Output Oa f16.
// Full tiles: K and V staged by async DMA.  Row-sum of P computed with an
// extra WMMA against an all-ones B fragment (no cross-lane reduction).
// ---------------------------------------------------------------------------
__global__ __launch_bounds__(128) void attn_kernel(
    const _Float16* __restrict__ Qh, const _Float16* __restrict__ Kc,
    const _Float16* __restrict__ Vt, _Float16* __restrict__ O,
    const float* __restrict__ bias_tab, int Nq, int Lk, int Lkp, int causal)
{
  __shared__ __attribute__((aligned(16))) _Float16 Ksh[32][72];    // [key][d]
  __shared__ __attribute__((aligned(16))) _Float16 Vsh[64][40];    // [d][key]
  __shared__ __attribute__((aligned(16))) _Float16 Psh[4][16][40]; // per-wave P
  const int tid = threadIdx.x;
  const int wave = tid >> 5, lane = tid & 31;
  const int l16 = lane & 15, lhi = lane >> 4;
  const int qb = blockIdx.x * 64 + wave * 16;
  const int b = blockIdx.y >> 3, h = blockIdx.y & 7;

  // Q fragments: each (chunk, group) is 8 contiguous halves = one 16B load
  v16h qf[2];
  {
    const _Float16* qrow = Qh + ((size_t)(b * Nq + qb + l16)) * IDM + h * DHH;
    #pragma unroll
    for (int c = 0; c < 2; ++c)
      #pragma unroll
      for (int g = 0; g < 2; ++g) {
        v8h qv = *(const v8h*)&qrow[c * 32 + g * 16 + lhi * 8];
        #pragma unroll
        for (int w = 0; w < 8; ++w) qf[c][g * 8 + w] = qv[w];
      }
  }
  v16h ones16;
  #pragma unroll
  for (int e = 0; e < 16; ++e) ones16[e] = (_Float16)1.0f;

  float m[8], lsum[8];
  v8f oacc[4];
  #pragma unroll
  for (int v = 0; v < 8; ++v) { m[v] = -3.0e38f; lsum[v] = 0.0f; }
  #pragma unroll
  for (int tt = 0; tt < 4; ++tt) oacc[tt] = zero8();

  int kmax = causal ? (blockIdx.x * 64 + 65) : Lk;   // row r sees keys <= r+1
  if (kmax > Lk) kmax = Lk;
  const int ntiles = (kmax + 31) >> 5;

  for (int t0 = 0; t0 < ntiles; ++t0) {
    const int key0 = t0 << 5;
    if (key0 + 32 <= Lk) {
      // full tile: K rows (32x64) and Vt rows (64x32) both via async DMA
      #pragma unroll
      for (int j = 0; j < 2; ++j) {
        int c = tid + 128 * j;          // 256 chunks for K
        int kk = c >> 3, d8 = c & 7;
        async_copy16(&Ksh[kk][d8 * 8],
                     &Kc[((size_t)b * Lk + key0 + kk) * DHH + d8 * 8]);
      }
      #pragma unroll
      for (int j = 0; j < 2; ++j) {
        int c = tid + 128 * j;          // 256 chunks for V (64 d x 4 chunks)
        int d = c >> 2, k8 = c & 3;
        async_copy16(&Vsh[d][k8 * 8],
                     &Vt[((size_t)b * DHH + d) * Lkp + key0 + k8 * 8]);
      }
    } else {
      // ragged tail tile (at most one per block): guarded scalar
      #pragma unroll
      for (int j = 0; j < 16; ++j) {
        int i = tid + 128 * j;
        int kk = i >> 6, d = i & 63;
        int key = key0 + kk;
        bool ok = (key < Lk);
        Ksh[kk][d] = ok ? Kc[((size_t)b * Lk + key) * DHH + d] : (_Float16)0.0f;
        Vsh[d][kk] = ok ? Vt[((size_t)b * DHH + d) * Lkp + key] : (_Float16)0.0f;
      }
    }
    wait_async0();
    __syncthreads();

    // preload all K fragments, then two S tiles (loads batch, waits stagger)
    v16h bk[2][2];
    #pragma unroll
    for (int kt = 0; kt < 2; ++kt) {
      #pragma unroll
      for (int e = 0; e < 16; ++e) bk[kt][0][e] = Ksh[kt * 16 + l16][(e >> 3) * 16 + lhi * 8 + (e & 7)];
      #pragma unroll
      for (int e = 0; e < 16; ++e) bk[kt][1][e] = Ksh[kt * 16 + l16][32 + (e >> 3) * 16 + lhi * 8 + (e & 7)];
    }
    v8f s[2];
    #pragma unroll
    for (int kt = 0; kt < 2; ++kt) {
      v8f z = wmma_f16(qf[0], bk[kt][0], zero8());
      s[kt]  = wmma_f16(qf[1], bk[kt][1], z);
    }

    // mask + relative position bias
    #pragma unroll
    for (int kt = 0; kt < 2; ++kt)
      #pragma unroll
      for (int v = 0; v < 8; ++v) {
        int i = qb + lhi * 8 + v;          // sim row
        int j = key0 + kt * 16 + l16;      // sim col (incl. null at 0)
        float sv = s[kt][v];
        if (j >= Lk || (causal && j > i + 1)) sv = -3.0e38f;
        else if (bias_tab) {
          int dd = i - j; if (dd < 0) dd = 0;
          sv += bias_tab[(h << 10) + dd];
        }
        s[kt][v] = sv;
      }

    // row max across the 16-lane half-groups: step-outer so 8 independent
    // permutes issue per step instead of one serial chain per row
    float mx[8];
    #pragma unroll
    for (int v = 0; v < 8; ++v) mx[v] = fmaxf(s[0][v], s[1][v]);
    #pragma unroll
    for (int off = 1; off < 16; off <<= 1)
      #pragma unroll
      for (int v = 0; v < 8; ++v)
        mx[v] = fmaxf(mx[v], __shfl_xor(mx[v], off, 16));

    float alpha[8];
    #pragma unroll
    for (int v = 0; v < 8; ++v) {
      float mnew = fmaxf(m[v], mx[v]);
      alpha[v] = __expf(m[v] - mnew);
      m[v] = mnew;
      Psh[wave][lhi * 8 + v][l16]      = (_Float16)__expf(s[0][v] - mnew);
      Psh[wave][lhi * 8 + v][16 + l16] = (_Float16)__expf(s[1][v] - mnew);
    }
    // Psh is wave-private and per-wave LDS ops are in-order: no barrier needed

    // preload P fragment + all 4 V fragments
    v16h pf, vf[4];
    #pragma unroll
    for (int e = 0; e < 16; ++e)
      pf[e] = Psh[wave][l16][lhi * 16 + e];
    #pragma unroll
    for (int tt = 0; tt < 4; ++tt)
      #pragma unroll
      for (int e = 0; e < 16; ++e) vf[tt][e] = Vsh[tt * 16 + l16][lhi * 16 + e];

    // row-sum of P via WMMA against all-ones B (replaces 32 ds_bpermutes)
    v8f rs = wmma_f16(pf, ones16, zero8());
    #pragma unroll
    for (int v = 0; v < 8; ++v) lsum[v] = lsum[v] * alpha[v] + rs[v];

    // O(16x64) = O*alpha + P(16x32) @ V_tile(32x64)
    #pragma unroll
    for (int tt = 0; tt < 4; ++tt) {
      #pragma unroll
      for (int e = 0; e < 8; ++e) oacc[tt][e] *= alpha[e];
      oacc[tt] = wmma_f16(pf, vf[tt], oacc[tt]);
    }
    __syncthreads();   // all reads of Ksh/Vsh done before next staging
  }

  #pragma unroll
  for (int tt = 0; tt < 4; ++tt) {
    int d = tt * 16 + l16;
    #pragma unroll
    for (int v = 0; v < 8; ++v) {
      int i = qb + lhi * 8 + v;
      O[((size_t)(b * Nq + i)) * IDM + h * DHH + d] =
          (_Float16)(oacc[tt][v] / lsum[v]);
    }
  }
}

// ---------------------------------------------------------------------------
// LayerNorm (one row per block, cols <= 512): out = LN(in)*g [+ resid];
// writes f16 (outh) when given, else f32 (outf). Optional stable max-div.
// ---------------------------------------------------------------------------
__global__ __launch_bounds__(256) void layernorm_kernel(
    const float* __restrict__ in, const float* __restrict__ g,
    const float* __restrict__ resid, float* __restrict__ outf,
    _Float16* __restrict__ outh, int cols, int stable)
{
  const int t = threadIdx.x;
  const size_t row = blockIdx.x;
  const float* x = in + row * cols;
  __shared__ float red[256];
  const bool a0 = (t < cols), a1 = (t + 256 < cols);
  float x0 = a0 ? x[t] : 0.0f;
  float x1 = a1 ? x[t + 256] : 0.0f;
  if (stable) {
    red[t] = fmaxf(a0 ? x0 : -3.0e38f, a1 ? x1 : -3.0e38f);
    __syncthreads();
    for (int s = 128; s > 0; s >>= 1) { if (t < s) red[t] = fmaxf(red[t], red[t + s]); __syncthreads(); }
    float rmax = red[0];
    __syncthreads();
    x0 /= rmax; x1 /= rmax;
  }
  red[t] = x0 + x1; __syncthreads();
  for (int s = 128; s > 0; s >>= 1) { if (t < s) red[t] += red[t + s]; __syncthreads(); }
  float mu = red[0] / (float)cols;
  __syncthreads();
  float d0 = a0 ? (x0 - mu) : 0.0f;
  float d1 = a1 ? (x1 - mu) : 0.0f;
  red[t] = d0 * d0 + d1 * d1; __syncthreads();
  for (int s = 128; s > 0; s >>= 1) { if (t < s) red[t] += red[t + s]; __syncthreads(); }
  float rstd = rsqrtf(red[0] / (float)cols + 1e-5f);
  float r0 = d0 * rstd * (a0 ? g[t] : 0.0f)       + ((resid && a0) ? resid[row * cols + t]       : 0.0f);
  float r1 = d1 * rstd * (a1 ? g[t + 256] : 0.0f) + ((resid && a1) ? resid[row * cols + t + 256] : 0.0f);
  if (outh) {
    if (a0) outh[row * cols + t]       = (_Float16)r0;
    if (a1) outh[row * cols + t + 256] = (_Float16)r1;
  } else {
    if (a0) outf[row * cols + t]       = r0;
    if (a1) outf[row * cols + t + 256] = r1;
  }
}

// ---------------------------------------------------------------------------
// qh <- f16( rotary(q) * DH^-0.5 ).  q:(B*Nq, 512) f32 = (b,n,h,d).
// ---------------------------------------------------------------------------
__global__ __launch_bounds__(256) void rotscale_q_kernel(
    const float* __restrict__ q, _Float16* __restrict__ qh, int Nq, int do_rot)
{
  const size_t row = blockIdx.x;            // b*Nq + n
  const int n = (int)(row % (size_t)Nq);
  const int t = threadIdx.x;
  const int hh = t >> 5, j = t & 31;
  const float* qp = q + row * IDM + hh * DHH;
  _Float16* qo = qh + row * IDM + hh * DHH;
  const float sc = 0.125f;                  // 64^-0.5
  if (do_rot) {
    if (j < 16) {
      float inv = powf(10000.0f, -(float)(2 * j) / 32.0f);
      float f = (float)n * inv;
      float c = __cosf(f), sn = __sinf(f);
      float a = qp[j], bb = qp[j + 16];
      qo[j]      = (_Float16)((a * c - bb * sn) * sc);
      qo[j + 16] = (_Float16)((bb * c + a * sn) * sc);
    } else {
      int d = 32 + (j - 16) * 2;
      qo[d]     = (_Float16)(qp[d] * sc);
      qo[d + 1] = (_Float16)(qp[d + 1] * sc);
    }
  } else {
    qo[2 * j]     = (_Float16)(qp[2 * j] * sc);
    qo[2 * j + 1] = (_Float16)(qp[2 * j + 1] * sc);
  }
}

// ---------------------------------------------------------------------------
// Kc:(B, Ltok+1, 64) and Vt:(B, 64, Lkp) f16 from kv:(B*Ltok,128) + null_kv.
// ---------------------------------------------------------------------------
__global__ __launch_bounds__(64) void build_kv_kernel(
    const float* __restrict__ kv, const float* __restrict__ nullkv,
    _Float16* __restrict__ Kc, _Float16* __restrict__ Vt,
    int Ltok, int Lkp, int do_rot)
{
  const int idx = blockIdx.x;               // 0 .. B*(Ltok+1)-1
  const int b = idx / (Ltok + 1);
  const int n = idx % (Ltok + 1);
  const int d = threadIdx.x;                // 0..63
  float kval, vval;
  if (n == 0) {
    kval = nullkv[d];
    vval = nullkv[DHH + d];
  } else {
    const float* src = kv + ((size_t)b * Ltok + (n - 1)) * 128;
    float kd = src[d];
    vval = src[DHH + d];
    if (do_rot && d < 32) {
      int jj = d & 15;
      float inv = powf(10000.0f, -(float)(2 * jj) / 32.0f);
      float f = (float)(n - 1) * inv;
      float c = __cosf(f), sn = __sinf(f);
      float other = src[(d < 16) ? d + 16 : d - 16];
      kval = (d < 16) ? (kd * c - other * sn) : (kd * c + other * sn);
    } else {
      kval = kd;
    }
  }
  Kc[((size_t)b * (Ltok + 1) + n) * DHH + d] = (_Float16)kval;
  Vt[((size_t)b * DHH + d) * Lkp + n]        = (_Float16)vval;
}

// ---------------------------------------------------------------------------
// bias_tab[h*1024 + n] = rel_emb[bucket(n)][h], rel_emb:(32, 8)
// ---------------------------------------------------------------------------
__global__ __launch_bounds__(256) void bias_table_kernel(
    const float* __restrict__ rel_emb, float* __restrict__ tab)
{
  int i = blockIdx.x * 256 + threadIdx.x;   // 0 .. 8*1024-1
  int hh = i >> 10, n = i & 1023;
  int bucket;
  if (n < 16) bucket = n;
  else {
    int large = 16 + (int)(logf((float)n / 16.0f) / logf(8.0f) * 16.0f);
    bucket = large < 31 ? large : 31;
  }
  tab[i] = rel_emb[bucket * HHD + hh];
}

// ---------------------------------------------------------------------------
// hsw[r, c] = f16( h1[r, c] * silu(h1[r, FI + c]) );  h1:(rows, 2*FI) f32
// ---------------------------------------------------------------------------
__global__ __launch_bounds__(256) void swiglu_kernel(
    const float* __restrict__ h1, _Float16* __restrict__ hsw, size_t total)
{
  size_t i = (size_t)blockIdx.x * 256 + threadIdx.x;
  if (i >= total) return;
  size_t r = i / FFI;
  int c = (int)(i % FFI);
  float hv = h1[r * (2 * FFI) + c];
  float gv = h1[r * (2 * FFI) + FFI + c];
  hsw[i] = (_Float16)(hv * (gv / (1.0f + __expf(-gv))));
}

__global__ __launch_bounds__(256) void add_kernel(
    float* __restrict__ x, const float* __restrict__ y, size_t n)
{
  size_t i = (size_t)blockIdx.x * 256 + threadIdx.x;
  if (i < n) x[i] += y[i];
}

// ---------------------------------------------------------------------------
// Weight convert + transpose: out[l][n][k] = f16(in[l][k][n]).
// Tiled 32x32 through LDS; coalesced on both sides.  K, Nc multiples of 32.
// grid = (Nc/32, K/32, L), block = 256.
// ---------------------------------------------------------------------------
__global__ __launch_bounds__(256) void cvt_transpose_kernel(
    const float* __restrict__ in, _Float16* __restrict__ out, int K, int Nc)
{
  __shared__ _Float16 tile[32][33];
  const size_t l = blockIdx.z;
  const float* src = in + l * (size_t)K * Nc;
  _Float16* dst = out + l * (size_t)K * Nc;
  const int n0 = blockIdx.x * 32, k0 = blockIdx.y * 32;
  const int tx = threadIdx.x & 31, ty = threadIdx.x >> 5;   // ty: 0..7
  #pragma unroll
  for (int i = 0; i < 4; ++i) {
    int k = ty + i * 8;
    tile[k][tx] = (_Float16)src[(size_t)(k0 + k) * Nc + n0 + tx];
  }
  __syncthreads();
  #pragma unroll
  for (int i = 0; i < 4; ++i) {
    int n = ty + i * 8;
    dst[(size_t)(n0 + n) * K + k0 + tx] = tile[tx][n];
  }
}

// ---------------------------------------------------------------------------
// Host orchestration
// ---------------------------------------------------------------------------
extern "C" void kernel_launch(void* const* d_in, const int* in_sizes, int n_in,
                              void* d_out, int out_size, void* d_ws, size_t ws_size,
                              hipStream_t stream)
{
  (void)in_sizes; (void)n_in; (void)out_size; (void)ws_size;
  const float* in_x       = (const float*)d_in[0];
  const float* in_ctx     = (const float*)d_in[1];
  const float* rel_emb    = (const float*)d_in[2];
  const float* sa_norm_g  = (const float*)d_in[3];
  const float* sa_wq      = (const float*)d_in[4];
  const float* sa_wkv     = (const float*)d_in[5];
  const float* sa_null_kv = (const float*)d_in[6];
  const float* sa_wo      = (const float*)d_in[7];
  const float* sa_out_g   = (const float*)d_in[8];
  const float* ca_norm_g  = (const float*)d_in[9];
  const float* ca_ctx_g   = (const float*)d_in[10];
  const float* ca_wq      = (const float*)d_in[11];
  const float* ca_wkv     = (const float*)d_in[12];
  const float* ca_null_kv = (const float*)d_in[13];
  const float* ca_wo      = (const float*)d_in[14];
  const float* ca_out_g   = (const float*)d_in[15];
  const float* ff_norm_g  = (const float*)d_in[16];
  const float* ff_w1      = (const float*)d_in[17];
  const float* ff_w2      = (const float*)d_in[18];
  const float* norm_g     = (const float*)d_in[19];

  float* ws = (float*)d_ws;
  size_t off = 0;
  auto carve = [&](size_t nfloats) {           // f32 slots, 256B aligned
    float* p = ws + off;
    off += (nfloats + 63) & ~((size_t)63);
    return p;
  };
  auto carve_h = [&](size_t nhalves) {         // f16 slots
    return (_Float16*)carve((nhalves + 1) / 2);
  };
  const size_t NX = (size_t)BB * NN * DD;      // 2M
  const int LKP_SA = 1056;   // (N+1=1025) padded to mult of 32
  const int LKP_CA = 544;    // (M+1=513)  padded to mult of 32

  // f32 buffers
  float* x    = carve(NX);
  float* q    = carve((size_t)BB * NN * IDM);
  float* kv   = carve((size_t)BB * NN * 128);
  float* proj = carve(NX);
  float* h1   = carve((size_t)BB * NN * 2 * FFI);
  float* bias = carve((size_t)HHD * 1024);
  // f16 activation buffers
  _Float16* xn16   = carve_h(NX);
  _Float16* ctxn16 = carve_h((size_t)BB * MMCT * PPC);
  _Float16* qh     = carve_h((size_t)BB * NN * IDM);
  _Float16* Kc16   = carve_h((size_t)BB * (NN + 1) * DHH);
  _Float16* Vt16   = carve_h((size_t)BB * DHH * LKP_SA);
  _Float16* Oa16   = carve_h((size_t)BB * NN * IDM);
  _Float16* hsw16  = carve_h((size_t)BB * NN * FFI);
  // f16 TRANSPOSED weight copies (converted once per launch)
  const size_t SWQ  = (size_t)LLY * DD * IDM;
  const size_t SWKV = (size_t)LLY * DD * 128;
  const size_t SWO  = (size_t)LLY * IDM * DD;
  const size_t CWKV = (size_t)LLY * PPC * 128;
  const size_t W1   = (size_t)LLY * DD * 2 * FFI;
  const size_t W2   = (size_t)LLY * FFI * DD;
  _Float16* sa_wqT  = carve_h(SWQ);
  _Float16* sa_wkvT = carve_h(SWKV);
  _Float16* sa_woT  = carve_h(SWO);
  _Float16* ca_wqT  = carve_h(SWQ);
  _Float16* ca_wkvT = carve_h(CWKV);
  _Float16* ca_woT  = carve_h(SWO);
  _Float16* ff_w1T  = carve_h(W1);
  _Float16* ff_w2T  = carve_h(W2);

  const int ROWS  = BB * NN;     // 4096
  const int CROWS = BB * MMCT;   // 2048

  hipMemcpyAsync(x, in_x, NX * sizeof(float), hipMemcpyDeviceToDevice, stream);
  bias_table_kernel<<<32, 256, 0, stream>>>(rel_emb, bias);

  auto cvtT = [&](const float* src, _Float16* dst, int K, int Nc) {
    cvt_transpose_kernel<<<dim3(Nc / 32, K / 32, LLY), 256, 0, stream>>>(src, dst, K, Nc);
  };
  cvtT(sa_wq,  sa_wqT,  DD, IDM);   cvtT(sa_wkv, sa_wkvT, DD, 128);
  cvtT(sa_wo,  sa_woT,  IDM, DD);   cvtT(ca_wq,  ca_wqT,  DD, IDM);
  cvtT(ca_wkv, ca_wkvT, PPC, 128);  cvtT(ca_wo,  ca_woT,  IDM, DD);
  cvtT(ff_w1,  ff_w1T,  DD, 2 * FFI); cvtT(ff_w2, ff_w2T, FFI, DD);

  for (int l = 0; l < LLY; ++l) {
    // ---------------- self attention ----------------
    layernorm_kernel<<<ROWS, 256, 0, stream>>>(x, sa_norm_g + l * DD, nullptr, nullptr, xn16, DD, 0);
    gemm_kernel<IDM><<<dim3(IDM / GBN, ROWS / GBM), 256, 0, stream>>>(
        xn16, sa_wqT + (size_t)l * DD * IDM, q, ROWS, DD);
    gemm_kernel<128><<<dim3(1, ROWS / GBM), 256, 0, stream>>>(
        xn16, sa_wkvT + (size_t)l * DD * 128, kv, ROWS, DD);
    rotscale_q_kernel<<<ROWS, 256, 0, stream>>>(q, qh, NN, 1);
    build_kv_kernel<<<BB * (NN + 1), 64, 0, stream>>>(
        kv, sa_null_kv + (size_t)l * 2 * DHH, Kc16, Vt16, NN, LKP_SA, 1);
    attn_kernel<<<dim3(NN / 64, BB * HHD), 128, 0, stream>>>(
        qh, Kc16, Vt16, Oa16, bias, NN, NN + 1, LKP_SA, 1);
    gemm_kernel<DD><<<dim3(DD / GBN, ROWS / GBM), 256, 0, stream>>>(
        Oa16, sa_woT + (size_t)l * IDM * DD, proj, ROWS, IDM);
    layernorm_kernel<<<ROWS, 256, 0, stream>>>(proj, sa_out_g + l * DD, x, x, nullptr, DD, 0);

    // ---------------- cross attention ----------------
    layernorm_kernel<<<ROWS, 256, 0, stream>>>(x, ca_norm_g + l * DD, nullptr, nullptr, xn16, DD, 0);
    layernorm_kernel<<<CROWS, 256, 0, stream>>>(in_ctx, ca_ctx_g + l * PPC, nullptr, nullptr, ctxn16, PPC, 0);
    gemm_kernel<IDM><<<dim3(IDM / GBN, ROWS / GBM), 256, 0, stream>>>(
        xn16, ca_wqT + (size_t)l * DD * IDM, q, ROWS, DD);
    gemm_kernel<128><<<dim3(1, CROWS / GBM), 256, 0, stream>>>(
        ctxn16, ca_wkvT + (size_t)l * PPC * 128, kv, CROWS, PPC);
    rotscale_q_kernel<<<ROWS, 256, 0, stream>>>(q, qh, NN, 0);
    build_kv_kernel<<<BB * (MMCT + 1), 64, 0, stream>>>(
        kv, ca_null_kv + (size_t)l * 2 * DHH, Kc16, Vt16, MMCT, LKP_CA, 0);
    attn_kernel<<<dim3(NN / 64, BB * HHD), 128, 0, stream>>>(
        qh, Kc16, Vt16, Oa16, nullptr, NN, MMCT + 1, LKP_CA, 0);
    gemm_kernel<DD><<<dim3(DD / GBN, ROWS / GBM), 256, 0, stream>>>(
        Oa16, ca_woT + (size_t)l * IDM * DD, proj, ROWS, IDM);
    layernorm_kernel<<<ROWS, 256, 0, stream>>>(proj, ca_out_g + l * DD, x, x, nullptr, DD, 0);

    // ---------------- feed forward (SwiGLU) ----------------
    layernorm_kernel<<<ROWS, 256, 0, stream>>>(x, ff_norm_g + l * DD, nullptr, nullptr, xn16, DD, 0);
    gemm_kernel<2 * FFI><<<dim3((2 * FFI) / GBN, ROWS / GBM), 256, 0, stream>>>(
        xn16, ff_w1T + (size_t)l * DD * 2 * FFI, h1, ROWS, DD);
    {
      size_t total = (size_t)ROWS * FFI;
      swiglu_kernel<<<(unsigned)((total + 255) / 256), 256, 0, stream>>>(h1, hsw16, total);
    }
    gemm_kernel<DD><<<dim3(DD / GBN, ROWS / GBM), 256, 0, stream>>>(
        hsw16, ff_w2T + (size_t)l * FFI * DD, proj, ROWS, FFI);
    add_kernel<<<(unsigned)((NX + 255) / 256), 256, 0, stream>>>(x, proj, NX);
  }

  // final stable layer norm -> d_out (f32)
  layernorm_kernel<<<ROWS, 256, 0, stream>>>(x, norm_g, nullptr, (float*)d_out, nullptr, DD, 1);
}
